// GATRegression_13597866459796
// MI455X (gfx1250) — compile-verified, matrix operand order checked by reading, not run on hardware
//
#include <hip/hip_runtime.h>
#include <cmath>

#define NNODES 50000
#define NEDGES 1600000
#define ETOT   (NEDGES + NNODES)
#define HIDCH  64
#define NHEADS 2

typedef __attribute__((ext_vector_type(16))) _Float16 v16h;
typedef __attribute__((ext_vector_type(8)))  _Float16 v8h;
typedef __attribute__((ext_vector_type(8)))  float    v8f;

// ---------- atomics ----------
__device__ __forceinline__ void atomic_max_f32(float* addr, float val) {
  // monotone int/uint trick; correct with -inf initialization
  if (!(val < 0.0f)) {
    atomicMax((int*)addr, __float_as_int(val));
  } else {
    atomicMin((unsigned int*)addr, __float_as_uint(val));
  }
}

__device__ __forceinline__ void atomic_add_f32(float* addr, float val) {
  __hip_atomic_fetch_add(addr, val, __ATOMIC_RELAXED, __HIP_MEMORY_SCOPE_AGENT);
}

// ---------- f32 -> f16 linear convert (A operand prep) ----------
__global__ void cvt_f16(const float* __restrict__ in, _Float16* __restrict__ out, int n) {
  int i = blockIdx.x * blockDim.x + threadIdx.x;
  if (i < n) out[i] = (_Float16)in[i];
}

// ---------- pack W[K,Ncols] (f32 row-major) into WMMA B-fragment order (f16) ----------
// frag f = ks*(Ncols/16)+tn holds 32 lanes x 16 halves contiguously:
//   Bp[(f*32 + lane)*16 + e] = W[k, tn*16 + (lane&15)]
//   k = ks*32 + (lane>>4)*8 + ((e>>3)<<4) + (e&7)
__global__ void pack_b(const float* __restrict__ W, _Float16* __restrict__ Bp,
                       int K, int Ncols) {
  int i = blockIdx.x * blockDim.x + threadIdx.x;
  if (i >= K * Ncols) return;
  int e    = i & 15;
  int lane = (i >> 4) & 31;
  int f    = i >> 9;
  int ntn  = Ncols >> 4;
  int ks   = f / ntn;
  int tn   = f - ks * ntn;
  int k    = ks * 32 + (lane >> 4) * 8 + ((e >> 3) << 4) + (e & 7);
  int col  = tn * 16 + (lane & 15);
  Bp[i] = (_Float16)W[k * Ncols + col];
}

// ---------- GEMM: C[M,NCOLS] = A[M,K](f16) @ Bp(packed f16), f32 accum ----------
// one wave per 16-row strip: A fragment loaded once per k-step, reused over all
// NCOLS/16 column tiles -> K/32 * NCOLS/16 back-to-back v_wmma per wave.
template <int K, int NCOLS>
__global__ void gemm_f16_wmma(const _Float16* __restrict__ A,
                              const _Float16* __restrict__ Bp,
                              float* __restrict__ C, int nwaves) {
  int gid  = blockIdx.x * blockDim.x + threadIdx.x;
  int wv   = gid >> 5;
  if (wv >= nwaves) return;              // whole-wave exit: EXEC all-ones inside
  int lane = threadIdx.x & 31;
  int half = lane >> 4;
  int l15  = lane & 15;
  constexpr int NTN = NCOLS / 16;

  const _Float16* arow = A + (size_t)(wv * 16 + l15) * K;

  v8f acc[NTN];
#pragma unroll
  for (int tn = 0; tn < NTN; ++tn) acc[tn] = (v8f){};

#pragma unroll
  for (int ks = 0; ks < K / 32; ++ks) {
    int kb = ks * 32 + half * 8;
    v8h alo = *(const v8h*)(arow + kb);        // 16B aligned contiguous
    v8h ahi = *(const v8h*)(arow + kb + 16);
    v16h af;
#pragma unroll
    for (int e = 0; e < 8; ++e) { af[e] = alo[e]; af[e + 8] = ahi[e]; }
#pragma unroll
    for (int tn = 0; tn < NTN; ++tn) {
      v16h bf = *(const v16h*)(Bp + ((size_t)(ks * NTN + tn) * 32 + lane) * 16);
      acc[tn] = __builtin_amdgcn_wmma_f32_16x16x32_f16(false, af, false, bf,
                                                       (short)0, acc[tn], false, false);
    }
  }
  // C/D layout: VGPR r -> row r + 8*half, col = lane&15
#pragma unroll
  for (int tn = 0; tn < NTN; ++tn) {
    float* crow = C + (size_t)(wv * 16 + half * 8) * NCOLS + tn * 16 + l15;
#pragma unroll
    for (int r = 0; r < 8; ++r) crow[(size_t)r * NCOLS] = acc[tn][r];
  }
}

// ---------- per-(node,head) attention logit halves: wave reduction ----------
__global__ void att_logits(const float* __restrict__ H, const float* __restrict__ a_src,
                           const float* __restrict__ a_dst, float* __restrict__ als,
                           float* __restrict__ ald, int n, int heads, int outc) {
  int gid  = blockIdx.x * blockDim.x + threadIdx.x;
  int wv   = gid >> 5;
  int lane = threadIdx.x & 31;
  if (wv >= n * heads) return;
  int node = wv / heads;
  int hd   = wv - node * heads;
  const float* hrow = H + (size_t)node * heads * outc + (size_t)hd * outc;
  const float* as = a_src + (size_t)hd * outc;
  const float* ad = a_dst + (size_t)hd * outc;
  float ss = 0.0f, sd = 0.0f;
  for (int c = lane; c < outc; c += 32) {
    float hv = hrow[c];
    ss += hv * as[c];
    sd += hv * ad[c];
  }
#pragma unroll
  for (int off = 16; off > 0; off >>= 1) {
    ss += __shfl_down(ss, off, 32);
    sd += __shfl_down(sd, off, 32);
  }
  if (lane == 0) { als[wv] = ss; ald[wv] = sd; }
}

// ---------- init: agg = 0, m = -inf, den = 0 ----------
__global__ void seg_init(float* __restrict__ agg, int nagg,
                         float* __restrict__ m, float* __restrict__ den, int nm) {
  int i = blockIdx.x * blockDim.x + threadIdx.x;
  if (i < nagg) agg[i] = 0.0f;
  if (i < nm) { m[i] = -__builtin_inff(); den[i] = 0.0f; }
}

// ---------- edge pass 1: leaky_relu logits + segment max ----------
__global__ void edge_logit_max(const int* __restrict__ ei, const float* __restrict__ als,
                               const float* __restrict__ ald, float* __restrict__ ex,
                               float* __restrict__ m, int heads) {
  int i = blockIdx.x * blockDim.x + threadIdx.x;
  if (i >= ETOT) return;
  int s, d;
  if (i < NEDGES) { s = ei[i]; d = ei[NEDGES + i]; } else { s = d = i - NEDGES; }
  for (int hd = 0; hd < heads; ++hd) {
    float l = als[s * heads + hd] + ald[d * heads + hd];
    l = l > 0.0f ? l : 0.2f * l;                 // LeakyReLU(0.2)
    ex[(size_t)i * heads + hd] = l;
    atomic_max_f32(&m[d * heads + hd], l);
  }
}

// ---------- edge pass 2: exp(l - max) + segment sum ----------
__global__ void edge_exp_sum(const int* __restrict__ ei, float* __restrict__ ex,
                             const float* __restrict__ m, float* __restrict__ den,
                             int heads) {
  int i = blockIdx.x * blockDim.x + threadIdx.x;
  if (i >= ETOT) return;
  int d = (i < NEDGES) ? ei[NEDGES + i] : (i - NEDGES);
  for (int hd = 0; hd < heads; ++hd) {
    float e = expf(ex[(size_t)i * heads + hd] - m[d * heads + hd]);
    ex[(size_t)i * heads + hd] = e;
    atomic_add_f32(&den[d * heads + hd], e);
  }
}

// ---------- edge pass 3: wave-per-edge weighted scatter-add ----------
__global__ void edge_scatter(const int* __restrict__ ei, const float* __restrict__ ex,
                             const float* __restrict__ den, const float* __restrict__ H,
                             float* __restrict__ agg, int heads, int outc) {
  int gid  = blockIdx.x * blockDim.x + threadIdx.x;
  int wv   = gid >> 5;
  int lane = threadIdx.x & 31;
  if (wv >= ETOT) return;
  int s, d;
  if (wv < NEDGES) { s = ei[wv]; d = ei[NEDGES + wv]; } else { s = d = wv - NEDGES; }
  int C = heads * outc;
  const float* hrow = H + (size_t)s * C;
  float* arow = agg + (size_t)d * C;
  for (int c = lane; c < C; c += 32) {
    int hd = c / outc;
    float alpha = ex[(size_t)wv * heads + hd] / (den[d * heads + hd] + 1e-16f);
    atomic_add_f32(&arow[c], hrow[c] * alpha);
  }
}

// ---------- layer-1 finalize: x2 = elu(agg + b) emitted directly as f16 GEMM operand ----------
__global__ void bias_elu_f16(const float* __restrict__ agg, const float* __restrict__ b,
                             _Float16* __restrict__ x2, int total, int C) {
  int i = blockIdx.x * blockDim.x + threadIdx.x;
  if (i >= total) return;
  int c = i % C;
  float v = agg[i] + b[c];
  x2[i] = (_Float16)(v > 0.0f ? v : expm1f(v));
}

// ---------- layer-2 finalize + regression head: out[n] = (agg+b2) . lin_w + lin_b ----------
__global__ void head_out(const float* __restrict__ agg, const float* __restrict__ b2,
                         const float* __restrict__ lw, const float* __restrict__ lb,
                         float* __restrict__ out, int n, int C) {
  int gid  = blockIdx.x * blockDim.x + threadIdx.x;
  int wv   = gid >> 5;
  int lane = threadIdx.x & 31;
  if (wv >= n) return;
  const float* arow = agg + (size_t)wv * C;
  float s = 0.0f;
  for (int c = lane; c < C; c += 32) s += (arow[c] + b2[c]) * lw[c];
#pragma unroll
  for (int off = 16; off > 0; off >>= 1) s += __shfl_down(s, off, 32);
  if (lane == 0) out[wv] = s + lb[0];
}

extern "C" void kernel_launch(void* const* d_in, const int* in_sizes, int n_in,
                              void* d_out, int out_size, void* d_ws, size_t ws_size,
                              hipStream_t stream) {
  const float* x     = (const float*)d_in[0];
  const int*   ei    = (const int*)  d_in[1];
  const float* W1    = (const float*)d_in[2];
  const float* asrc1 = (const float*)d_in[3];
  const float* adst1 = (const float*)d_in[4];
  const float* b1    = (const float*)d_in[5];
  const float* W2    = (const float*)d_in[6];
  const float* asrc2 = (const float*)d_in[7];
  const float* adst2 = (const float*)d_in[8];
  const float* b2    = (const float*)d_in[9];
  const float* linw  = (const float*)d_in[10];
  const float* linb  = (const float*)d_in[11];
  float* out = (float*)d_out;

  // ---- workspace layout (float units; f16 regions counted as halves/2) ----
  float* ws = (float*)d_ws;
  _Float16* Xf16  = (_Float16*)ws;                         // N*128 halves = N*64 floats
  _Float16* X2f16 = (_Float16*)(ws + (size_t)NNODES * 64); // N*128 halves
  _Float16* Bp1   = (_Float16*)(ws + (size_t)NNODES * 128);          // 128*128 halves = 8192 fl
  _Float16* Bp2   = (_Float16*)(ws + (size_t)NNODES * 128 + 8192);   // 128*64 halves = 4096 fl
  float* H1   = ws + (size_t)NNODES * 128 + 16384;         // N*128 f32
  float* AGG1 = H1 + (size_t)NNODES * 128;                 // N*128 f32
  float* EX   = AGG1 + (size_t)NNODES * 128;               // ETOT*2
  float* als1 = EX + (size_t)ETOT * 2;                     // N*2
  float* ald1 = als1 + NNODES * NHEADS;
  float* m1   = ald1 + NNODES * NHEADS;
  float* den1 = m1 + NNODES * NHEADS;
  float* als2 = den1 + NNODES * NHEADS;                    // N
  float* ald2 = als2 + NNODES;
  float* m2   = ald2 + NNODES;
  float* den2 = m2 + NNODES;
  float* H2   = H1;                                        // N*64 (H1 dead by then)
  float* AGG2 = H1 + (size_t)NNODES * 64;                  // N*64

  const int TB = 256;
  const int RT = NNODES / 16;                              // 3125 row tiles (exact)

  // ---- operand prep ----
  cvt_f16<<<(NNODES * 128 + TB - 1) / TB, TB, 0, stream>>>(x, Xf16, NNODES * 128);
  pack_b<<<(128 * 128 + TB - 1) / TB, TB, 0, stream>>>(W1, Bp1, 128, 128);
  pack_b<<<(128 * 64 + TB - 1) / TB, TB, 0, stream>>>(W2, Bp2, 128, 64);

  // ===== layer 1: GATConv(128 -> 64, heads=2, concat) =====
  {
    gemm_f16_wmma<128, 128><<<(RT * 32 + 127) / 128, 128, 0, stream>>>(Xf16, Bp1, H1, RT);

    int lw = NNODES * NHEADS;
    att_logits<<<(lw * 32 + 127) / 128, 128, 0, stream>>>(H1, asrc1, adst1, als1, ald1,
                                                          NNODES, NHEADS, HIDCH);
    int nagg = NNODES * 128;
    seg_init<<<(nagg + TB - 1) / TB, TB, 0, stream>>>(AGG1, nagg, m1, den1, NNODES * NHEADS);

    edge_logit_max<<<(ETOT + TB - 1) / TB, TB, 0, stream>>>(ei, als1, ald1, EX, m1, NHEADS);
    edge_exp_sum  <<<(ETOT + TB - 1) / TB, TB, 0, stream>>>(ei, EX, m1, den1, NHEADS);
    long sthr = (long)ETOT * 32;
    edge_scatter<<<(int)((sthr + TB - 1) / TB), TB, 0, stream>>>(ei, EX, den1, H1, AGG1,
                                                                 NHEADS, HIDCH);
    bias_elu_f16<<<(nagg + TB - 1) / TB, TB, 0, stream>>>(AGG1, b1, X2f16, nagg, 128);
  }

  // ===== layer 2: GATConv(128 -> 64, heads=1) =====
  {
    gemm_f16_wmma<128, 64><<<(RT * 32 + 127) / 128, 128, 0, stream>>>(X2f16, Bp2, H2, RT);

    att_logits<<<(NNODES * 32 + 127) / 128, 128, 0, stream>>>(H2, asrc2, adst2, als2, ald2,
                                                              NNODES, 1, HIDCH);
    int nagg = NNODES * 64;
    seg_init<<<(nagg + TB - 1) / TB, TB, 0, stream>>>(AGG2, nagg, m2, den2, NNODES);

    edge_logit_max<<<(ETOT + TB - 1) / TB, TB, 0, stream>>>(ei, als2, ald2, EX, m2, 1);
    edge_exp_sum  <<<(ETOT + TB - 1) / TB, TB, 0, stream>>>(ei, EX, m2, den2, 1);
    long sthr = (long)ETOT * 32;
    edge_scatter<<<(int)((sthr + TB - 1) / TB), TB, 0, stream>>>(ei, EX, den2, H2, AGG2,
                                                                 1, HIDCH);
    head_out<<<(NNODES * 32 + TB - 1) / TB, TB, 0, stream>>>(AGG2, b2, linw, linb, out,
                                                             NNODES, HIDCH);
  }
}
